// NL_Conv3_72481868087447
// MI455X (gfx1250) — matrix operand order repaired and checked by reading, not compiled, not run on hardware
//
#include <hip/hip_runtime.h>
#include <hip/hip_bf16.h>

typedef __attribute__((ext_vector_type(16))) _Float16 v16h;
typedef __attribute__((ext_vector_type(8)))  float    v8f;
typedef __attribute__((ext_vector_type(4)))  float    v4f;
typedef __attribute__((ext_vector_type(4)))  unsigned int u32x4;
typedef __attribute__((ext_vector_type(8)))  int      i32x8;
typedef __attribute__((ext_vector_type(4)))  int      i32x4;

#define NL_EPS 1e-4f
#define APK_HALFS (144 * 5 * 32 * 16)     // A packed: [tile][kstep][lane][16 halves]
#define O_HALFS   (9 * 5 * 32 * 16)       // one output-channel slab = 23040 halves
#define AO_BYTES  (O_HALFS * 2)           // 46080 B per o
#define AO_DWORDS (AO_BYTES / 4)          // 11520
#define XC_BYTES  (4 * 2304 * 4)          // 36864 B (4 waves)
#define SMEM_BYTES (XC_BYTES + 2 * AO_BYTES)   // 129024 B

#if __has_builtin(__builtin_amdgcn_tensor_load_to_lds) && \
    __has_builtin(__builtin_amdgcn_s_wait_tensorcnt)
#define NL_HAS_TDM 1
#else
#define NL_HAS_TDM 0
#endif

// ---------------------------------------------------------------------------
// Pack conv weights [2304,16,3,3] f32 -> WMMA A-operand layout, f16.
// k = (kh*3+kw)*16 + ic, padded to 160. A 16x32 k order used here (contiguous, B matches):
//   lanes 0-15 : M=lane,    halves j=0..15 -> K=j
//   lanes 16-31: M=lane-16, halves j=0..15 -> K=16+j
// ---------------------------------------------------------------------------
__global__ __launch_bounds__(256) void nl_pack_a(const float* __restrict__ cw,
                                                 _Float16* __restrict__ apk) {
    int idx = blockIdx.x * 256 + threadIdx.x;          // < 368640
    int j  = idx & 15;
    int l  = (idx >> 4) & 31;
    int s  = (idx >> 9) % 5;
    int t  = idx / 2560;
    int oc = t * 16 + (l & 15);
    int hi = l >> 4;
    int kl = hi * 16 + j;
    int kg = s * 32 + kl;
    float v = 0.0f;
    if (kg < 144) {
        int ic   = kg & 15;
        int kpos = kg >> 4;                            // kh*3+kw
        v = cw[(oc * 16 + ic) * 9 + kpos];             // [oc][ic][kh][kw]
    }
    apk[idx] = (_Float16)v;
}

// x [8,16,64,64] f32 (NCHW) -> NHWC f16 so one (row,col) gives 16 contiguous ch.
__global__ __launch_bounds__(256) void nl_pack_x(const float* __restrict__ x,
                                                 _Float16* __restrict__ xt) {
    int idx = blockIdx.x * 256 + threadIdx.x;          // < 524288
    int ic = idx & 15;
    int w  = (idx >> 4) & 63;
    int h  = (idx >> 10) & 63;
    int b  = idx >> 16;
    xt[idx] = (_Float16)x[(((b * 16 + ic) * 64) + h) * 64 + w];
}

#if NL_HAS_TDM
// Issue one TDM DMA: 46080 contiguous bytes (one o-slab of A) global -> LDS.
// D# per CDNA5 ISA: group0 {count=1, lds_addr, global_addr,
// type=2}; group1 {data_size=4B, tensor_dim0=tile_dim0=11520, dims1..=1}.
__device__ __forceinline__ void nl_tdm_load(unsigned lds_off,
                                            const _Float16* gsrc) {
    unsigned long long ga = (unsigned long long)(uintptr_t)gsrc;
    u32x4 g0 = { 1u,                                   // count=1, user D#
                 lds_off,                               // lds_addr (bytes)
                 (unsigned)ga,                          // global_addr[31:0]
                 ((unsigned)(ga >> 32) & 0x01FFFFFFu) | (2u << 30) }; // type=2
    i32x8 g1 = { (int)(2u << 16),                      // data_size = 4 bytes
                 (int)((AO_DWORDS & 0xFFFFu) << 16),   // tensor_dim0[15:0]
                 (int)((AO_DWORDS >> 16) | (1u << 16)),// dim0[31:16] | tensor_dim1=1
                 (int)((unsigned)AO_DWORDS << 16),     // tile_dim0 = 11520
                 1,                                    // tile_dim1=1, tile_dim2=0
                 (int)AO_DWORDS,                       // tensor_dim0_stride lo32
                 0, 0 };
    i32x4 z4 = {0, 0, 0, 0};
#if __clang_major__ >= 23
    i32x8 z8 = {0, 0, 0, 0, 0, 0, 0, 0};
    __builtin_amdgcn_tensor_load_to_lds(g0, g1, z4, z4, z8, 0);
#else
    __builtin_amdgcn_tensor_load_to_lds(g0, g1, z4, z4, 0);
#endif
}
#endif

// ---------------------------------------------------------------------------
// Main fused kernel. Block = 128 threads = 4 waves = one (b,h) row; each wave
// a 16-pixel tile. A slabs are DMA'd per-o into double-buffered LDS by the
// TDM (issued by wave 0, overlapped with the 45 WMMAs of the previous o).
// Per p-block: clause of 10 ds_load_b128 (A for all 5 k-steps), one DS wait,
// then 5 back-to-back v_wmma_f32_16x16x32_f16 (accumulator-chained only),
// then ReLU+eps folding into per-o numerator/denominator with xc from LDS.
// ---------------------------------------------------------------------------
__global__ __launch_bounds__(128) void nl_main(const float* __restrict__ x,
                                               const _Float16* __restrict__ apk,
                                               const _Float16* __restrict__ xt,
                                               float* __restrict__ out) {
    extern __shared__ char smem[];
    const int wave = threadIdx.x >> 5;
    const int lane = threadIdx.x & 31;
    const int b  = blockIdx.x >> 6;
    const int h  = blockIdx.x & 63;
    const int w0 = wave * 16;
    const int px = lane & 15;
    const int hi = lane >> 4;
    float* xcw = (float*)smem + wave * 2304;

    // xc tiles: element (p = pb*16 + 8*hi + v, px) at [((pb*2+hi)*16+px)*8+v].
    // xc[p,px] = x[b, p%16, (h-j)&63, (w-i)&63], shift idx p/16 = (i+1)*3+(j+1).
    for (int idx = lane; idx < 2304; idx += 32) {
        int v   = idx & 7;
        int pxl = (idx >> 3) & 15;
        int hib = (idx >> 7) & 1;
        int pb  = idx >> 8;
        int p   = pb * 16 + hib * 8 + v;
        int c   = p & 15;
        int sh  = p >> 4;
        int si  = sh / 3 - 1;
        int sj  = sh % 3 - 1;
        int row = (h - sj) & 63;
        int col = (w0 + pxl - si) & 63;
        xcw[idx] = x[(((b * 16 + c) * 64) + row) * 64 + col];
    }

    // B operand (f16 32x16): lane = column N=l%16, lanes>=16 hold K=16..31.
    // Step s, half j -> k = s*32 + hi*16 + j = 16 contiguous channels at
    // neighborhood position q = 2s+hi (zero-padded conv borders; q==9 = K pad).
    v16h Bm[5];
#pragma unroll
    for (int s = 0; s < 5; ++s) {
        int q = 2 * s + hi;
        v16h bb = {};
        if (q < 9) {
            int kh = q / 3, kw = q % 3;
            int row = h + kh - 1;
            int col = w0 + px + kw - 1;
            if (row >= 0 && row < 64 && col >= 0 && col < 64)
                bb = *(const v16h*)(xt + (((b * 64 + row) * 64) + col) * 16);
        }
        Bm[s] = bb;
    }

#if NL_HAS_TDM
    if (wave == 0)                          // prologue DMA: A slab for o=0
        nl_tdm_load((unsigned)(uintptr_t)(smem + XC_BYTES), apk);
#endif
    __syncthreads();                        // xc ready

    for (int o = 0; o < 16; ++o) {
        const _Float16* abuf =
            (const _Float16*)(smem + XC_BYTES + (o & 1) * AO_BYTES);
#if NL_HAS_TDM
        if (wave == 0) __builtin_amdgcn_s_wait_tensorcnt(0);
        __syncthreads();                    // publish A(o); prior readers done
        if (o < 15 && wave == 0)            // overlap DMA(o+1) with compute(o)
            nl_tdm_load((unsigned)(uintptr_t)
                            (smem + XC_BYTES + ((o + 1) & 1) * AO_BYTES),
                        apk + (o + 1) * O_HALFS);
#else
        {   // fallback: cooperative synchronous stage (no TDM builtin)
            const unsigned* src = (const unsigned*)(apk + o * O_HALFS);
            unsigned* dst = (unsigned*)(smem + XC_BYTES);
            for (int i = threadIdx.x; i < AO_DWORDS; i += 128) dst[i] = src[i];
            __syncthreads();
            abuf = (const _Float16*)(smem + XC_BYTES);
        }
#endif

        float num = 0.0f, den = 0.0f;
#pragma unroll 1
        for (int pb = 0; pb < 9; ++pb) {
            // Phase 1: stage all 5 A operands (one ds clause, one wait).
            v16h a[5];
#pragma unroll
            for (int s = 0; s < 5; ++s)
                a[s] = *(const v16h*)(abuf + ((pb * 5 + s) * 32 + lane) * 16);
            // Phase 1b: stage the 8 xc values for this p-block too.
            const v4f* xcp = (const v4f*)(xcw + (((pb * 2 + hi) * 16) + px) * 8);
            v4f xlo = xcp[0], xhi = xcp[1];
            // Phase 2: 5 back-to-back WMMAs (chained through the accumulator).
            v8f d = {};
#pragma unroll
            for (int s = 0; s < 5; ++s)
                d = __builtin_amdgcn_wmma_f32_16x16x32_f16(
                        false, a[s], false, Bm[s], (short)0, d, false, false);
            // Phase 3: ReLU+eps fold into numerator/denominator.
#pragma unroll
            for (int v = 0; v < 8; ++v) {
                float wij = fmaxf(d[v], 0.0f) + NL_EPS;
                float xv  = (v < 4) ? xlo[v & 3] : xhi[v & 3];
                den += wij;
                num = fmaf(wij, xv, num);
            }
        }
        den += __shfl_xor(den, 16, 32);     // combine M-halves (lanes l, l^16)
        num += __shfl_xor(num, 16, 32);
        if (hi == 0)
            out[(((b * 16 + o) * 64) + h) * 64 + w0 + px] = num / den;

#if !NL_HAS_TDM
        __syncthreads();                    // single buffer: guard overwrite
#endif
    }
}

extern "C" void kernel_launch(void* const* d_in, const int* in_sizes, int n_in,
                              void* d_out, int out_size, void* d_ws, size_t ws_size,
                              hipStream_t stream) {
    const float* x  = (const float*)d_in[0];   // [8,16,64,64] f32
    const float* cw = (const float*)d_in[1];   // [2304,16,3,3] f32
    float* out = (float*)d_out;                // [8,16,64,64] f32

    _Float16* apk = (_Float16*)d_ws;                                         // 737280 B
    _Float16* xt  = (_Float16*)((char*)d_ws + APK_HALFS * sizeof(_Float16)); // 1 MB

    nl_pack_a<<<1440, 256, 0, stream>>>(cw, apk);
    nl_pack_x<<<2048, 256, 0, stream>>>(x, xt);
    nl_main<<<512, 128, SMEM_BYTES, stream>>>(x, apk, xt, out);
}
